// MoE_69406671503752
// MI455X (gfx1250) — matrix operand order repaired and checked by reading, not compiled, not run on hardware
//
#include <hip/hip_runtime.h>
#include <hip/hip_bf16.h>

// ---------------------------------------------------------------------------
// MoE forward for MI455X (gfx1250, wave32, WMMA).
// Heavy GEMMs: bf16 operands, f32 accumulate via v_wmma_f32_16x16x32_bf16,
// 128x128 workgroup tile, double-buffered LDS, async global->LDS staging of
// the activation operand (ASYNCcnt), VALU-converted transposed weight tiles.
// Router / LayerNorm stats / losses: fp32 VALU.
// ---------------------------------------------------------------------------

typedef __bf16 v16bf __attribute__((ext_vector_type(16)));
typedef float  v8f   __attribute__((ext_vector_type(8)));

#define TOK   8192          // 256 * 32
#define HID   768
#define FFN_D 3072
#define NEXP  8
#define NCLS  191
#define KC    32            // WMMA K chunk
#define MT    128           // workgroup M tile
#define NT    128           // workgroup N tile

// ---- gfx1250 async global->LDS copy (16B per lane), ASYNCcnt tracked ------
__device__ __forceinline__ void async_cp_b128(const __bf16* gsrc, __bf16* ldst) {
    unsigned lds_off = (unsigned)(uintptr_t)ldst;             // low 32b = LDS addr
    unsigned long long ga = (unsigned long long)(uintptr_t)gsrc;
    asm volatile("global_load_async_to_lds_b128 %0, %1, off"
                 :: "v"(lds_off), "v"(ga) : "memory");
}
__device__ __forceinline__ void wait_async0() {
    asm volatile("s_wait_asynccnt 0x0" ::: "memory");
}

// ---------------------------------------------------------------------------
__global__ void zero_f32(float* __restrict__ p, int n) {
    int i = blockIdx.x * blockDim.x + threadIdx.x;
    for (; i < n; i += gridDim.x * blockDim.x) p[i] = 0.f;
}

__global__ void cvt_f32_bf16(const float* __restrict__ x, __bf16* __restrict__ y, int n) {
    int i = blockIdx.x * blockDim.x + threadIdx.x;
    for (; i < n; i += gridDim.x * blockDim.x) y[i] = (__bf16)x[i];
}

// ---------------------------------------------------------------------------
// Tiled WMMA GEMM: C[M,N] = A_bf16[M,K] @ B_f32[K,N] + bias
//   mode 0: out_bf16[m*N+n] = bf16(acc + bias[n])           (pre-LN buffer)
//   mode 1: out_f32[m*N+n] += (acc + bias[n]) * mask[m*ms]  (expert scatter)
// 256 threads = 8 waves; tile 128(M) x 128(N); wave -> 32 x 64 strip
// (2 x 4 WMMA tiles). K staged in 32-chunks, double-buffered LDS; A tile via
// async global->LDS b128, B tile converted f32->bf16 and transposed to [n][k].
// M, K multiples of 128 / 32 (true here); N bounds-guarded (191 case).
// ---------------------------------------------------------------------------
__global__ __launch_bounds__(256)
void gemm_wmma_bf16(const __bf16* __restrict__ A, int lda,
                    const float*  __restrict__ B, int ldb,
                    const float*  __restrict__ bias,
                    int M, int N, int K, int mode,
                    __bf16* __restrict__ outB,
                    float*  __restrict__ outF,
                    const float* __restrict__ maskBase, int maskStride)
{
    __shared__ __bf16 As[2][MT][KC];   // [buf][m][k]
    __shared__ __bf16 Bs[2][NT][KC];   // [buf][n][k] (transposed during fill)

    const int tid  = threadIdx.x;
    const int lane = tid & 31;
    const int wave = tid >> 5;
    const int wm   = wave & 3;      // 0..3 -> M sub-tile (32 rows each)
    const int wn   = wave >> 2;     // 0..1 -> N sub-tile (64 cols each)
    const int l15  = lane & 15;
    const int lh   = lane >> 4;     // 0 or 1

    const int m0 = blockIdx.y * MT;
    const int n0 = blockIdx.x * NT;

    // A fill: thread -> row rA, 16 contiguous bf16 at cA split into 2 async b128
    const int rA = tid >> 1;            // 0..127
    const int cA = (tid & 1) * 16;      // 0 or 16
    // B fill: thread -> k-row kr, 16 consecutive n
    const int kr = tid >> 3;            // 0..31
    const int nc = (tid & 7) * 16;      // 0..112
    const bool vecOK = ((ldb & 3) == 0) && (n0 + NT <= N);

    v8f acc[2][4];
#pragma unroll
    for (int i = 0; i < 2; ++i)
#pragma unroll
        for (int j = 0; j < 4; ++j) acc[i][j] = (v8f){};

    auto fillA = [&](int buf, int k0) {
        const __bf16* g = &A[(size_t)(m0 + rA) * lda + k0 + cA];
        async_cp_b128(g,     &As[buf][rA][cA]);
        async_cp_b128(g + 8, &As[buf][rA][cA + 8]);
    };
    auto fillB = [&](int buf, int k0) {
        const float* Brow = B + (size_t)(k0 + kr) * ldb;
        if (vecOK) {
#pragma unroll
            for (int q = 0; q < 4; ++q) {
                float4 f = *(const float4*)&Brow[n0 + nc + q * 4];
                Bs[buf][nc + q * 4 + 0][kr] = (__bf16)f.x;
                Bs[buf][nc + q * 4 + 1][kr] = (__bf16)f.y;
                Bs[buf][nc + q * 4 + 2][kr] = (__bf16)f.z;
                Bs[buf][nc + q * 4 + 3][kr] = (__bf16)f.w;
            }
        } else {
#pragma unroll
            for (int i = 0; i < 16; ++i) {
                int n = n0 + nc + i;
                float v = (n < N) ? Brow[n] : 0.f;
                Bs[buf][nc + i][kr] = (__bf16)v;
            }
        }
    };

    union Frag { uint4 u[2]; v16bf v; };

    int buf = 0;
    fillA(0, 0);
    fillB(0, 0);
    wait_async0();
    __syncthreads();

    for (int k0 = 0; k0 < K; k0 += KC) {
        const int nb = buf ^ 1;
        if (k0 + KC < K) {
            fillA(nb, k0 + KC);                 // async: overlaps WMMAs below
            fillB(nb, k0 + KC);
            __builtin_prefetch(&B[(size_t)(k0 + 2 * KC + kr) * ldb + n0 + nc], 0, 1);
        }

        // ---- fragments per documented 16-bit A/B VGPR layouts ----
        Frag a[2], b[4];
#pragma unroll
        for (int i = 0; i < 2; ++i) {
            const __bf16* ap = &As[buf][wm * 32 + i * 16 + l15][lh * 8];
            a[i].u[0] = *(const uint4*)ap;          // K[lh*8 .. +7]
            a[i].u[1] = *(const uint4*)(ap + 16);   // K[16+lh*8 .. +7]
        }
#pragma unroll
        for (int j = 0; j < 4; ++j) {
            const __bf16* bp = &Bs[buf][wn * 64 + j * 16 + l15][lh * 8];
            b[j].u[0] = *(const uint4*)bp;          // K[lh*8 .. +7]
            b[j].u[1] = *(const uint4*)(bp + 16);   // K[16+lh*8 .. +7]
        }
#pragma unroll
        for (int i = 0; i < 2; ++i)
#pragma unroll
            for (int j = 0; j < 4; ++j)
                acc[i][j] = __builtin_amdgcn_wmma_f32_16x16x32_bf16(
                    false, a[i].v, false, b[j].v, (short)0, acc[i][j], false, false);

        wait_async0();      // prefetched chunk landed (had whole compute phase)
        __syncthreads();
        buf = nb;
    }

    // ---- epilogue: lanes 0-15: N=lane, M=r; lanes 16-31: N=lane-16, M=8+r --
#pragma unroll
    for (int i = 0; i < 2; ++i) {
#pragma unroll
        for (int r = 0; r < 8; ++r) {
            const int m = m0 + wm * 32 + i * 16 + r + lh * 8;
#pragma unroll
            for (int j = 0; j < 4; ++j) {
                const int n = n0 + wn * 64 + j * 16 + l15;
                if (n < N) {
                    float v = acc[i][j][r] + bias[n];
                    if (mode == 0) outB[(size_t)m * N + n] = (__bf16)v;
                    else outF[(size_t)m * N + n] += v * maskBase[(size_t)m * maskStride];
                }
            }
        }
    }
}

// ---------------------------------------------------------------------------
// Row LayerNorm + ReLU (bf16 in -> bf16 out), one block per row.
// ---------------------------------------------------------------------------
__global__ __launch_bounds__(256)
void ln_relu_bf16(const __bf16* __restrict__ X,
                  const float* __restrict__ g, const float* __restrict__ b,
                  __bf16* __restrict__ Y, int D)
{
    const int row = blockIdx.x;
    const __bf16* x = X + (size_t)row * D;
    __shared__ float s1[256], s2[256];
    float a = 0.f, q = 0.f;
    for (int i = threadIdx.x; i < D; i += 256) {
        float v = (float)x[i];
        a += v; q += v * v;
    }
    s1[threadIdx.x] = a; s2[threadIdx.x] = q;
    __syncthreads();
    for (int s = 128; s > 0; s >>= 1) {
        if (threadIdx.x < s) {
            s1[threadIdx.x] += s1[threadIdx.x + s];
            s2[threadIdx.x] += s2[threadIdx.x + s];
        }
        __syncthreads();
    }
    const float inv = 1.f / (float)D;
    const float mean = s1[0] * inv;
    const float var  = s2[0] * inv - mean * mean;
    const float r    = rsqrtf(var + 1e-5f);
    __bf16* y = Y + (size_t)row * D;
    for (int i = threadIdx.x; i < D; i += 256) {
        float v = ((float)x[i] - mean) * r * g[i] + b[i];
        y[i] = (__bf16)(v > 0.f ? v : 0.f);
    }
}

// ---------------------------------------------------------------------------
// Router: logits (fp32), softmax, top-2 normalized mask, loss accumulators.
// ---------------------------------------------------------------------------
__global__ __launch_bounds__(256)
void router_kernel(const __bf16* __restrict__ hb,
                   const float* __restrict__ rw2, const float* __restrict__ rb2,
                   float* __restrict__ mask,
                   float* __restrict__ zsum, float* __restrict__ loadsum, int T)
{
    int t = blockIdx.x * blockDim.x + threadIdx.x;
    if (t >= T) return;
    float acc[NEXP];
#pragma unroll
    for (int e = 0; e < NEXP; ++e) acc[e] = rb2[e];
    const __bf16* h = hb + (size_t)t * HID;
    for (int k = 0; k < HID; ++k) {
        float hv = (float)h[k];
        const float* w = rw2 + (size_t)k * NEXP;
#pragma unroll
        for (int e = 0; e < NEXP; ++e) acc[e] += hv * w[e];
    }
    float mx = acc[0];
#pragma unroll
    for (int e = 1; e < NEXP; ++e) mx = fmaxf(mx, acc[e]);
    float p[NEXP], se = 0.f;
#pragma unroll
    for (int e = 0; e < NEXP; ++e) { p[e] = __expf(acc[e] - mx); se += p[e]; }
    float lse = mx + __logf(se);
    float isv = 1.f / se;
#pragma unroll
    for (int e = 0; e < NEXP; ++e) p[e] *= isv;
    int i1 = 0;
#pragma unroll
    for (int e = 1; e < NEXP; ++e) if (p[e] > p[i1]) i1 = e;
    int i2 = (i1 == 0) ? 1 : 0;
#pragma unroll
    for (int e = 0; e < NEXP; ++e) if (e != i1 && e != i2 && p[e] > p[i2]) i2 = e;
    float s2 = p[i1] + p[i2];
    float v1 = p[i1] / s2, v2 = p[i2] / s2;
    float* mrow = mask + (size_t)t * NEXP;
#pragma unroll
    for (int e = 0; e < NEXP; ++e) mrow[e] = 0.f;
    mrow[i1] = v1; mrow[i2] = v2;
    atomicAdd(zsum, lse * lse);
    atomicAdd(&loadsum[i1], v1);
    atomicAdd(&loadsum[i2], v2);
}

// ---------------------------------------------------------------------------
__global__ void mean_kernel(const float* __restrict__ fin, float* __restrict__ out) {
    int idx = blockIdx.x * blockDim.x + threadIdx.x;
    if (idx >= 256 * NCLS) return;
    int b = idx / NCLS, c = idx % NCLS;
    float s = 0.f;
#pragma unroll
    for (int n = 0; n < 32; ++n) s += fin[(size_t)(b * 32 + n) * NCLS + c];
    out[idx] = s * (1.f / 32.f);
}

__global__ void finalize_kernel(const float* __restrict__ zsum,
                                const float* __restrict__ loadsum,
                                float* __restrict__ out)
{
    float z = 0.1f * (zsum[0] / (float)TOK);
    float lb = 0.f;
#pragma unroll
    for (int e = 0; e < NEXP; ++e) {
        float l = loadsum[e] / (float)TOK - 0.125f;
        lb += l * l;
    }
    out[0] = z + 0.1f * lb;
}

// ---------------------------------------------------------------------------
// Host launcher
// ---------------------------------------------------------------------------
extern "C" void kernel_launch(void* const* d_in, const int* in_sizes, int n_in,
                              void* d_out, int out_size, void* d_ws, size_t ws_size,
                              hipStream_t stream) {
    const float* x    = (const float*)d_in[0];   // [256,32,768]
    const float* rw1  = (const float*)d_in[1];   // [768,768]
    const float* rb1  = (const float*)d_in[2];
    const float* rlg  = (const float*)d_in[3];
    const float* rlb  = (const float*)d_in[4];
    const float* rw2  = (const float*)d_in[5];   // [768,8]
    const float* rb2  = (const float*)d_in[6];
    const float* ew1  = (const float*)d_in[7];   // [8,768,3072]
    const float* eb1  = (const float*)d_in[8];
    const float* el1g = (const float*)d_in[9];
    const float* el1b = (const float*)d_in[10];
    const float* ew2  = (const float*)d_in[11];  // [8,3072,768]
    const float* eb2  = (const float*)d_in[12];
    const float* el2g = (const float*)d_in[13];
    const float* el2b = (const float*)d_in[14];
    const float* ew3  = (const float*)d_in[15];  // [8,768,191]
    const float* eb3  = (const float*)d_in[16];

    char*  ws  = (char*)d_ws;
    size_t off = 0;
    auto alloc = [&](size_t bytes) -> void* {
        void* p = ws + off;
        off += (bytes + 255) & ~(size_t)255;
        return p;
    };
    __bf16* xb     = (__bf16*)alloc((size_t)TOK * HID   * 2);  // x in bf16
    __bf16* preB   = (__bf16*)alloc((size_t)TOK * FFN_D * 2);  // pre-LN buffer
    __bf16* postB  = (__bf16*)alloc((size_t)TOK * FFN_D * 2);  // post-LN buffer
    float*  finAcc = (float*) alloc((size_t)TOK * NCLS  * 4);  // final logits acc
    float*  mask   = (float*) alloc((size_t)TOK * NEXP  * 4);
    float*  losses = (float*) alloc(9 * 4);                    // [zsum, load x8]
    float*  zsum    = losses;
    float*  loadsum = losses + 1;

    float* out = (float*)d_out;   // 256*191 logits + 1 aux

    zero_f32<<<(TOK * NCLS + 255) / 256, 256, 0, stream>>>(finAcc, TOK * NCLS);
    zero_f32<<<1, 32, 0, stream>>>(losses, 9);
    cvt_f32_bf16<<<(TOK * HID + 255) / 256, 256, 0, stream>>>(x, xb, TOK * HID);

    // router GEMM1 + LN/ReLU + logits/top-2/mask
    gemm_wmma_bf16<<<dim3(HID / NT, TOK / MT), 256, 0, stream>>>(
        xb, HID, rw1, HID, rb1, TOK, HID, HID, 0, preB, nullptr, nullptr, 0);
    ln_relu_bf16<<<TOK, 256, 0, stream>>>(preB, rlg, rlb, postB, HID);
    router_kernel<<<TOK / 256, 256, 0, stream>>>(postB, rw2, rb2, mask, zsum, loadsum, TOK);

    // experts (dense, mask-weighted scatter — identical semantics to ref)
    for (int e = 0; e < NEXP; ++e) {
        const float* w1 = ew1 + (size_t)e * HID * FFN_D;
        const float* b1 = eb1 + (size_t)e * FFN_D;
        const float* g1 = el1g + (size_t)e * FFN_D;
        const float* B1 = el1b + (size_t)e * FFN_D;
        const float* w2 = ew2 + (size_t)e * FFN_D * HID;
        const float* b2 = eb2 + (size_t)e * HID;
        const float* g2 = el2g + (size_t)e * HID;
        const float* B2 = el2b + (size_t)e * HID;
        const float* w3 = ew3 + (size_t)e * HID * NCLS;
        const float* b3 = eb3 + (size_t)e * NCLS;

        gemm_wmma_bf16<<<dim3(FFN_D / NT, TOK / MT), 256, 0, stream>>>(
            xb, HID, w1, FFN_D, b1, TOK, FFN_D, HID, 0, preB, nullptr, nullptr, 0);
        ln_relu_bf16<<<TOK, 256, 0, stream>>>(preB, g1, B1, postB, FFN_D);

        gemm_wmma_bf16<<<dim3(HID / NT, TOK / MT), 256, 0, stream>>>(
            postB, FFN_D, w2, HID, b2, TOK, HID, FFN_D, 0, preB, nullptr, nullptr, 0);
        ln_relu_bf16<<<TOK, 256, 0, stream>>>(preB, g2, B2, postB, HID);

        gemm_wmma_bf16<<<dim3((NCLS + NT - 1) / NT, TOK / MT), 256, 0, stream>>>(
            postB, HID, w3, NCLS, b3, TOK, NCLS, HID, 1,
            nullptr, finAcc, mask + e, NEXP);
    }

    mean_kernel<<<(256 * NCLS + 255) / 256, 256, 0, stream>>>(finAcc, out);
    finalize_kernel<<<1, 1, 0, stream>>>(zsum, loadsum, out + 256 * NCLS);
}
